// patch_head_73967926771912
// MI455X (gfx1250) — matrix operand order repaired and checked by reading, not compiled, not run on hardware
//
#include <hip/hip_runtime.h>
#include <hip/hip_bf16.h>
#include <math.h>
#include <stdint.h>

typedef __attribute__((ext_vector_type(16))) __bf16 v16bf;
typedef __attribute__((ext_vector_type(8)))  float  v8f;

#define BB    64
#define NN    196
#define DD    384
#define HH    6
#define HEADD 64
#define BN    (BB * NN)      // 12544
#define RR    (BN + BB)      // 12608 == 64*197
#define DFFF  1536
#define K2S   8
#define KNUMS 4
#define BROWB 80             // LDS B-tile row stride (bytes): 16B aligned, bank-conflict-free

// ---------------------------------------------------------------------------
// GEMM: D[M,N] = epilogue( A[M,K](bf16) x Bt[N,K](bf16)^T ), fp32 accumulate.
// Block 128 = 4 waves -> 64x64 tile; B tile staged in LDS via
// global_load_async_to_lds_b128 with double buffering (ASYNCcnt).
// M,N multiples of 64; K multiple of 32.
// Epilogue compile-time flags: (+bias[N]) -> (gelu?) -> (+Cadd) -> f32/bf16.
// ---------------------------------------------------------------------------
template <bool HB, bool HC, bool GEL, bool OF, bool OH>
__global__ void __launch_bounds__(128)
ph_gemm_bf16_wmma(const __bf16* __restrict__ A, const __bf16* __restrict__ Bt,
                  const float* __restrict__ Cadd, const float* __restrict__ bias,
                  float* __restrict__ outF, __bf16* __restrict__ outH,
                  int N, int K) {
  __shared__ __align__(16) unsigned char Bsm[2][64 * BROWB];

  int t    = threadIdx.x;
  int lane = t & 31;
  int wave = t >> 5;
  int half = lane >> 4;
  int l15  = lane & 15;
  int tileM = blockIdx.y * 64 + wave * 16;
  int tileN = blockIdx.x * 64;

  // 32-bit LDS byte offsets (generic LDS addr[31:0] == LDS offset)
  unsigned ldsb[2];
  ldsb[0] = (unsigned)(uintptr_t)&Bsm[0][0];
  ldsb[1] = (unsigned)(uintptr_t)&Bsm[1][0];

  // Async copy of one 64x32 bf16 B tile (4KB) into an LDS buffer:
  // 256 x 16B chunks over 128 threads -> 2 global_load_async_to_lds_b128 each.
  auto issueB = [&](int k0, unsigned base) {
#pragma unroll
    for (int c2 = 0; c2 < 2; ++c2) {
      int ch   = t * 2 + c2;     // 0..255
      int n    = ch >> 2;        // row within tile (0..63)
      int part = ch & 3;         // 16B chunk within the row's 64B of K-data
      const __bf16* g = Bt + (size_t)(tileN + n) * (size_t)K + (size_t)(k0 + part * 8);
      unsigned l = base + (unsigned)(n * BROWB + part * 16);
      asm volatile("global_load_async_to_lds_b128 %0, %1, off"
                   :: "v"(l), "v"((unsigned long long)(uintptr_t)g)
                   : "memory");
    }
  };

  v8f zero = {0.f, 0.f, 0.f, 0.f, 0.f, 0.f, 0.f, 0.f};
  v8f acc[4];
#pragma unroll
  for (int c = 0; c < 4; ++c) acc[c] = zero;

  // A-fragment base: lane row (tileM + l15); per K-step the fragment is two
  // 16B chunks at byte offsets 2*k0 + 32*half and +16 (both 16B aligned).
  const unsigned char* abase =
      (const unsigned char*)(A + (size_t)(tileM + l15) * (size_t)K) + (size_t)(32 * half);

  // Prologue: stage first B tile.
  issueB(0, ldsb[0]);
  asm volatile("s_wait_asynccnt 0x0" ::: "memory");
  __syncthreads();

  int nsteps = K >> 5;
  for (int s = 0; s < nsteps; ++s) {
    int cur = s & 1;

    // Overlap: issue next tile's async DMA before the WMMAs of this tile.
    if (s + 1 < nsteps) {
      issueB((s + 1) << 5, ldsb[cur ^ 1]);
      __builtin_prefetch(abase + (size_t)((s + 1) << 6), 0, 3);  // global_prefetch_b8
    }

    union { v16bf v; uint4 q[2]; } fa;
    {
      const unsigned char* p = abase + (size_t)(s << 6);
      fa.q[0] = *(const uint4*)(p);
      fa.q[1] = *(const uint4*)(p + 16);
    }

#pragma unroll
    for (int c = 0; c < 4; ++c) {
      // B frag: lane -> row n = c*16 + l15; 32 contiguous bytes at K = 16*half.
      const unsigned char* rowp = &Bsm[cur][(c * 16 + l15) * BROWB + half * 32];
      union { v16bf v; uint4 q[2]; } fb;
      fb.q[0] = *(const uint4*)(rowp);
      fb.q[1] = *(const uint4*)(rowp + 16);
      acc[c] = __builtin_amdgcn_wmma_f32_16x16x32_bf16(
          false, fa.v, false, fb.v, (short)0, acc[c], false, false);
    }

    if (s + 1 < nsteps) {
      asm volatile("s_wait_asynccnt 0x0" ::: "memory");
      __syncthreads();
    }
  }

#pragma unroll
  for (int c = 0; c < 4; ++c) {
    int col = tileN + c * 16 + l15;
#pragma unroll
    for (int r = 0; r < 8; ++r) {
      int row = tileM + half * 8 + r;
      size_t o = (size_t)row * (size_t)N + (size_t)col;
      float v = acc[c][r];
      if constexpr (HB)  v += bias[col];
      if constexpr (GEL) v = 0.5f * v * (1.0f + erff(v * 0.70710678118654752f));
      if constexpr (HC)  v += Cadd[o];
      if constexpr (OF)  outF[o] = v;
      if constexpr (OH)  outH[o] = (__bf16)v;
    }
  }
}

// ---------------------------------------------------------------------------
// Row LayerNorm (D=384) -> bf16. One block (128 threads) per row.
// ---------------------------------------------------------------------------
__global__ void __launch_bounds__(128)
ph_ln_rows_bf16(const float* __restrict__ X, const float* __restrict__ g,
                const float* __restrict__ bb, __bf16* __restrict__ Y) {
  __shared__ float sA[128], sB[128];
  __shared__ float s_mean, s_inv;
  int r = blockIdx.x, t = threadIdx.x;
  const float* xr = X + (size_t)r * DD;
  float x0 = xr[t], x1 = xr[t + 128], x2 = xr[t + 256];
  sA[t] = x0 + x1 + x2;
  sB[t] = x0 * x0 + x1 * x1 + x2 * x2;
  __syncthreads();
  for (int o = 64; o > 0; o >>= 1) {
    if (t < o) { sA[t] += sA[t + o]; sB[t] += sB[t + o]; }
    __syncthreads();
  }
  if (t == 0) {
    float m = sA[0] * (1.0f / DD);
    float var = sB[0] * (1.0f / DD) - m * m;
    s_mean = m;
    s_inv  = rsqrtf(var + 1e-6f);
  }
  __syncthreads();
  float m = s_mean, inv = s_inv;
  __bf16* yr = Y + (size_t)r * DD;
  yr[t]       = (__bf16)((x0 - m) * inv * g[t]       + bb[t]);
  yr[t + 128] = (__bf16)((x1 - m) * inv * g[t + 128] + bb[t + 128]);
  yr[t + 256] = (__bf16)((x2 - m) * inv * g[t + 256] + bb[t + 256]);
}

// Final LayerNorm: CLS rows -> fp32 output with (b, [cls | loc]) interleave.
__global__ void __launch_bounds__(128)
ph_ln_out_final(const float* __restrict__ CLS, const float* __restrict__ g,
                const float* __restrict__ bb, float* __restrict__ out) {
  __shared__ float sA[128], sB[128];
  __shared__ float s_mean, s_inv;
  int r = blockIdx.x, t = threadIdx.x;
  int bidx = r / 197, j = r % 197;
  int src = (j == 0) ? bidx : (BB + bidx * NN + (j - 1));
  const float* xr = CLS + (size_t)src * DD;
  float x0 = xr[t], x1 = xr[t + 128], x2 = xr[t + 256];
  sA[t] = x0 + x1 + x2;
  sB[t] = x0 * x0 + x1 * x1 + x2 * x2;
  __syncthreads();
  for (int o = 64; o > 0; o >>= 1) {
    if (t < o) { sA[t] += sA[t + o]; sB[t] += sB[t + o]; }
    __syncthreads();
  }
  if (t == 0) {
    float m = sA[0] * (1.0f / DD);
    float var = sB[0] * (1.0f / DD) - m * m;
    s_mean = m;
    s_inv  = rsqrtf(var + 1e-6f);
  }
  __syncthreads();
  float m = s_mean, inv = s_inv;
  float* yr = out + (size_t)r * DD;
  yr[t]       = (x0 - m) * inv * g[t]       + bb[t];
  yr[t + 128] = (x1 - m) * inv * g[t + 128] + bb[t + 128];
  yr[t + 256] = (x2 - m) * inv * g[t + 256] + bb[t + 256];
}

// ---------------------------------------------------------------------------
// Weight transpose + f32->bf16:  W[K][N] -> Wt[N][K]
// ---------------------------------------------------------------------------
__global__ void __launch_bounds__(256)
ph_transpose_bf16(const float* __restrict__ W, __bf16* __restrict__ Wt, int K, int N) {
  int i = blockIdx.x * 256 + threadIdx.x;
  int k = i / N, n = i % N;
  Wt[(size_t)n * K + k] = (__bf16)W[i];
}

__global__ void __launch_bounds__(256)
ph_init_cls(const float* __restrict__ cls_t, float* __restrict__ CLS) {
  int i = blockIdx.x * 256 + threadIdx.x;
  CLS[i] = cls_t[i % DD];
}

// ---------------------------------------------------------------------------
// Top-4 neighbor selection (cosine sim over 8 neighbors). One wave per (b,n).
// ---------------------------------------------------------------------------
__device__ inline float ph_wave_sum(float v) {
  for (int o = 16; o > 0; o >>= 1) v += __shfl_xor(v, o, 32);
  return v;
}

__global__ void __launch_bounds__(32)
ph_topk(const float* __restrict__ x, const int* __restrict__ lidx,
        int* __restrict__ fidx) {
  int s = blockIdx.x;
  int b = s / NN, n = s % NN;
  int lane = threadIdx.x;
  const float* xr = x + (size_t)s * DD;
  float cs = 0.f;
  for (int d = lane; d < DD; d += 32) { float v = xr[d]; cs += v * v; }
  cs = ph_wave_sum(cs);
  float cnorm = fmaxf(sqrtf(cs), 1e-12f);
  float sim[K2S];
  for (int k = 0; k < K2S; ++k) {
    int p = lidx[n * K2S + k];
    const float* nr = x + ((size_t)b * NN + p) * DD;
    float dt = 0.f, nn2 = 0.f;
    for (int d = lane; d < DD; d += 32) {
      float a = nr[d];
      dt += a * xr[d];
      nn2 += a * a;
    }
    dt  = ph_wave_sum(dt);
    nn2 = ph_wave_sum(nn2);
    sim[k] = dt / (cnorm * fmaxf(sqrtf(nn2), 1e-12f));
  }
  if (lane == 0) {
    unsigned used = 0;
    for (int t = 0; t < KNUMS; ++t) {
      int best = 0;
      float bv = -1e30f;
      for (int k = 0; k < K2S; ++k)
        if (!((used >> k) & 1u) && sim[k] > bv) { bv = sim[k]; best = k; }
      used |= 1u << best;
      fidx[s * KNUMS + t] = lidx[n * K2S + best];
    }
  }
}

// ---------------------------------------------------------------------------
// Attention, glo path: 197 keys (cls + 196 patches). Grid (64, 6), block 64.
// ---------------------------------------------------------------------------
__global__ void __launch_bounds__(64)
ph_attn_glo(const float* __restrict__ Qc, const float* __restrict__ Kc,
            const float* __restrict__ Kx, const float* __restrict__ Vc,
            const float* __restrict__ Vx, __bf16* __restrict__ O) {
  __shared__ float lg[197];
  __shared__ float s_inv;
  int b = blockIdx.x, h = blockIdx.y, t = threadIdx.x;
  const float* q = Qc + (size_t)b * DD + h * HEADD;
  for (int j = t; j < 197; j += 64) {
    const float* kr = (j == 0) ? (Kc + (size_t)b * DD + h * HEADD)
                               : (Kx + ((size_t)b * NN + (j - 1)) * DD + h * HEADD);
    float s = 0.f;
#pragma unroll 8
    for (int d = 0; d < HEADD; ++d) s += q[d] * kr[d];
    lg[j] = s * 0.125f;
  }
  __syncthreads();
  if (t == 0) {
    float mx = -1e30f;
    for (int j = 0; j < 197; ++j) mx = fmaxf(mx, lg[j]);
    float sm = 0.f;
    for (int j = 0; j < 197; ++j) { lg[j] = expf(lg[j] - mx); sm += lg[j]; }
    s_inv = 1.0f / sm;
  }
  __syncthreads();
  float inv = s_inv;
  float o = 0.f;
  for (int j = 0; j < 197; ++j) {
    const float* vr = (j == 0) ? (Vc + (size_t)b * DD + h * HEADD)
                               : (Vx + ((size_t)b * NN + (j - 1)) * DD + h * HEADD);
    o += lg[j] * vr[t];
  }
  O[(size_t)b * DD + h * HEADD + t] = (__bf16)(o * inv);
}

// ---------------------------------------------------------------------------
// Attention, loc path: 5 keys (cls + 4 gathered patches). Grid 12544, block 64.
// ---------------------------------------------------------------------------
__global__ void __launch_bounds__(64)
ph_attn_loc(const float* __restrict__ Qc, const float* __restrict__ Kc,
            const float* __restrict__ Kx, const float* __restrict__ Vc,
            const float* __restrict__ Vx, const int* __restrict__ fidx,
            __bf16* __restrict__ O) {
  __shared__ float lg[HH * 5];
  __shared__ float pr[HH * 5];
  __shared__ int   kr[5];
  int s = blockIdx.x;
  int b = s / NN;
  int t = threadIdx.x;
  if (t < 4) kr[t + 1] = b * NN + fidx[s * KNUMS + t];
  if (t == 4) kr[0] = 0;  // slot 0 handled specially below
  __syncthreads();
  size_t crow = (size_t)(BB + s) * DD;
  if (t < HH * 5) {
    int h = t / 5, j = t % 5;
    const float* q = Qc + crow + h * HEADD;
    const float* k = (j == 0) ? (Kc + crow + h * HEADD)
                              : (Kx + (size_t)kr[j] * DD + h * HEADD);
    float sdot = 0.f;
#pragma unroll 8
    for (int d = 0; d < HEADD; ++d) sdot += q[d] * k[d];
    lg[t] = sdot * 0.125f;
  }
  __syncthreads();
  if (t < HH) {
    float mx = -1e30f;
    for (int j = 0; j < 5; ++j) mx = fmaxf(mx, lg[t * 5 + j]);
    float sm = 0.f;
    float e[5];
    for (int j = 0; j < 5; ++j) { e[j] = expf(lg[t * 5 + j] - mx); sm += e[j]; }
    float inv = 1.0f / sm;
    for (int j = 0; j < 5; ++j) pr[t * 5 + j] = e[j] * inv;
  }
  __syncthreads();
#pragma unroll
  for (int h = 0; h < HH; ++h) {
    float o = 0.f;
    for (int j = 0; j < 5; ++j) {
      const float* vr = (j == 0) ? (Vc + crow + h * HEADD)
                                 : (Vx + (size_t)kr[j] * DD + h * HEADD);
      o += pr[h * 5 + j] * vr[t];
    }
    O[crow + h * HEADD + t] = (__bf16)o;
  }
}

// ---------------------------------------------------------------------------
// Host driver
// ---------------------------------------------------------------------------
extern "C" void kernel_launch(void* const* d_in, const int* in_sizes, int n_in,
                              void* d_out, int out_size, void* d_ws, size_t ws_size,
                              hipStream_t stream) {
  (void)in_sizes; (void)n_in; (void)out_size; (void)ws_size;
  const float* x     = (const float*)d_in[0];
  const int*   lidx  = (const int*)d_in[1];
  const float* cls_t = (const float*)d_in[2];
  const float* ng    = (const float*)d_in[3];
  const float* nb    = (const float*)d_in[4];
  const float* Wq    = (const float*)d_in[5];
  const float* Wk    = (const float*)d_in[6];
  const float* Wv    = (const float*)d_in[7];
  const float* Wo    = (const float*)d_in[8];
  const float* ln1g  = (const float*)d_in[9];
  const float* ln1b  = (const float*)d_in[10];
  const float* ln2g  = (const float*)d_in[11];
  const float* ln2b  = (const float*)d_in[12];
  const float* W1    = (const float*)d_in[13];
  const float* b1    = (const float*)d_in[14];
  const float* W2    = (const float*)d_in[15];
  const float* b2    = (const float*)d_in[16];

  char* ws = (char*)d_ws;
  size_t off = 0;
  auto alloc = [&](size_t bytes) -> void* {
    off = (off + 255) & ~(size_t)255;
    void* p = ws + off;
    off += bytes;
    return p;
  };

  __bf16 *wqt[2], *wkt[2], *wvt[2], *wot[2], *w1t[2], *w2t[2];
  for (int i = 0; i < 2; ++i) {
    wqt[i] = (__bf16*)alloc((size_t)DD * DD * 2);
    wkt[i] = (__bf16*)alloc((size_t)DD * DD * 2);
    wvt[i] = (__bf16*)alloc((size_t)DD * DD * 2);
    wot[i] = (__bf16*)alloc((size_t)DD * DD * 2);
    w1t[i] = (__bf16*)alloc((size_t)DD * DFFF * 2);
    w2t[i] = (__bf16*)alloc((size_t)DFFF * DD * 2);
  }
  int*    fidx  = (int*)alloc((size_t)BN * KNUMS * 4);
  float*  CLS   = (float*)alloc((size_t)RR * DD * 4);
  __bf16* xn    = (__bf16*)alloc((size_t)BN * DD * 2);
  __bf16* clsln = (__bf16*)alloc((size_t)RR * DD * 2);
  float*  Kx    = (float*)alloc((size_t)BN * DD * 4);
  float*  Vx    = (float*)alloc((size_t)BN * DD * 4);
  float*  Qc    = (float*)alloc((size_t)RR * DD * 4);
  float*  Kc    = (float*)alloc((size_t)RR * DD * 4);
  float*  Vc    = (float*)alloc((size_t)RR * DD * 4);
  __bf16* O     = (__bf16*)alloc((size_t)RR * DD * 2);
  __bf16* hidden = (__bf16*)Qc;  // alias: Qc+Kc (dead after attention) == RR*DFF*2 bytes
  __bf16* cn     = clsln;        // alias: clsln dead after Q/K/V projections

  // Weight conversion + transpose (once; covers both depths)
  for (int i = 0; i < 2; ++i) {
    size_t o44 = (size_t)i * DD * DD;
    ph_transpose_bf16<<<(DD * DD) / 256, 256, 0, stream>>>(Wq + o44, wqt[i], DD, DD);
    ph_transpose_bf16<<<(DD * DD) / 256, 256, 0, stream>>>(Wk + o44, wkt[i], DD, DD);
    ph_transpose_bf16<<<(DD * DD) / 256, 256, 0, stream>>>(Wv + o44, wvt[i], DD, DD);
    ph_transpose_bf16<<<(DD * DD) / 256, 256, 0, stream>>>(Wo + o44, wot[i], DD, DD);
    ph_transpose_bf16<<<(DD * DFFF) / 256, 256, 0, stream>>>(W1 + (size_t)i * DD * DFFF, w1t[i], DD, DFFF);
    ph_transpose_bf16<<<(DFFF * DD) / 256, 256, 0, stream>>>(W2 + (size_t)i * DFFF * DD, w2t[i], DFFF, DD);
  }

  ph_topk<<<BN, 32, 0, stream>>>(x, lidx, fidx);
  ph_init_cls<<<((size_t)RR * DD) / 256, 256, 0, stream>>>(cls_t, CLS);

  dim3 blk(128);
  dim3 g384(DD / 64, BN / 64);    // (6, 196)
  dim3 gR384(DD / 64, RR / 64);   // (6, 197)
  dim3 gFF(DFFF / 64, RR / 64);   // (24, 197)

  for (int i = 0; i < 2; ++i) {
    // LN1 of patch rows (shared glo/loc) and of all cls rows
    ph_ln_rows_bf16<<<BN, 128, 0, stream>>>(x, ln1g + i * DD, ln1b + i * DD, xn);
    ph_ln_rows_bf16<<<RR, 128, 0, stream>>>(CLS, ln1g + i * DD, ln1b + i * DD, clsln);

    // Projections: plain store-f32 GEMMs
    ph_gemm_bf16_wmma<false, false, false, true, false><<<g384, blk, 0, stream>>>(
        xn, wkt[i], nullptr, nullptr, Kx, nullptr, DD, DD);
    ph_gemm_bf16_wmma<false, false, false, true, false><<<g384, blk, 0, stream>>>(
        xn, wvt[i], nullptr, nullptr, Vx, nullptr, DD, DD);
    ph_gemm_bf16_wmma<false, false, false, true, false><<<gR384, blk, 0, stream>>>(
        clsln, wqt[i], nullptr, nullptr, Qc, nullptr, DD, DD);
    ph_gemm_bf16_wmma<false, false, false, true, false><<<gR384, blk, 0, stream>>>(
        clsln, wkt[i], nullptr, nullptr, Kc, nullptr, DD, DD);
    ph_gemm_bf16_wmma<false, false, false, true, false><<<gR384, blk, 0, stream>>>(
        clsln, wvt[i], nullptr, nullptr, Vc, nullptr, DD, DD);

    // Attention -> O (bf16), rows 0..63 glo, 64.. loc
    ph_attn_glo<<<dim3(BB, HH), 64, 0, stream>>>(Qc, Kc, Kx, Vc, Vx, O);
    ph_attn_loc<<<BN, 64, 0, stream>>>(Qc, Kc, Kx, Vc, Vx, fidx, O);

    // cls += O @ Wo   (C-accumulate)
    ph_gemm_bf16_wmma<false, true, false, true, false><<<gR384, blk, 0, stream>>>(
        O, wot[i], CLS, nullptr, CLS, nullptr, DD, DD);

    // FFN: hidden = gelu(LN2(cls) @ W1 + b1) [bf16 out]; cls += hidden @ W2 + b2
    ph_ln_rows_bf16<<<RR, 128, 0, stream>>>(CLS, ln2g + i * DD, ln2b + i * DD, cn);
    ph_gemm_bf16_wmma<true, false, true, false, true><<<gFF, blk, 0, stream>>>(
        cn, w1t[i], nullptr, b1 + (size_t)i * DFFF, nullptr, hidden, DFFF, DD);
    ph_gemm_bf16_wmma<true, true, false, true, false><<<gR384, blk, 0, stream>>>(
        hidden, w2t[i], CLS, b2 + (size_t)i * DD, CLS, nullptr, DD, DFFF);
  }

  ph_ln_out_final<<<BB * 197, 128, 0, stream>>>(CLS, ng, nb, (float*)d_out);
}
